// Attention_90744069030375
// MI455X (gfx1250) — matrix operand order repaired and checked by reading, not compiled, not run on hardware
//
#include <hip/hip_runtime.h>

typedef __bf16 v2bf  __attribute__((ext_vector_type(2)));
typedef __bf16 v8bf  __attribute__((ext_vector_type(8)));
typedef __bf16 v16bf __attribute__((ext_vector_type(16)));
typedef float  v8f   __attribute__((ext_vector_type(8)));
typedef float  v4f   __attribute__((ext_vector_type(4)));

#define B_  8
#define S_  2048
#define D_  1024
#define BM  32
#define BN  32
#define QPITCH  2064          // D*2 + 16B pad -> 4-bank rotation per row
#define KPITCH  2064
#define KTPITCH 80            // BN*2 + 16B pad
#define OFF_Q     0
#define OFF_K     66048       // 32*2064
#define OFF_KT    132096      // + 32*2064
#define OFF_SP    214016      // + 1024*80
#define OFF_P     222208      // + 2*32*32*4
#define OFF_ALPHA 224768      // + 32*80
#define OFF_LINV  224896
#define SMEM_BYTES 225024

static __device__ __forceinline__ unsigned short f2bf(float f) {
  __bf16 h = (__bf16)f;                       // RTE convert fp32 -> bf16
  return __builtin_bit_cast(unsigned short, h);
}
// pack two fp32 -> packed bf16x2 (one v_cvt_pk_bf16_f32 when available)
static __device__ __forceinline__ unsigned pk2(float x, float y) {
#if defined(__has_builtin)
#if __has_builtin(__builtin_amdgcn_cvt_pk_bf16_f32)
  v2bf r = __builtin_amdgcn_cvt_pk_bf16_f32(x, y);
  return __builtin_bit_cast(unsigned, r);
#else
  return (unsigned)f2bf(x) | ((unsigned)f2bf(y) << 16);
#endif
#else
  return (unsigned)f2bf(x) | ((unsigned)f2bf(y) << 16);
#endif
}
static __device__ __forceinline__ v16bf cat8(v8bf lo, v8bf hi) {
  return __builtin_shufflevector(lo, hi, 0,1,2,3,4,5,6,7,8,9,10,11,12,13,14,15);
}

__global__ __launch_bounds__(256, 1)
void fa_fwd_kernel(const float* __restrict__ qg, const float* __restrict__ kg,
                   const unsigned char* __restrict__ maskg,
                   const float* __restrict__ scaleg, float* __restrict__ outg) {
  __shared__ __align__(16) char smem[SMEM_BYTES];
  const int tid  = threadIdx.x;
  const int w    = tid >> 5;
  const int lane = tid & 31;
  const int lrow = lane & 15;
  const bool hih = lane >= 16;
  const int b  = blockIdx.y;
  const int q0 = blockIdx.x * BM;
  const float scaleV = scaleg[0];

  // ---- load Q tile (fp32 -> bf16, row-major, streamed/non-temporal) ----
  {
    const int row  = tid >> 3;
    const int csub = (tid & 7) * 4;
    const float* src = qg + ((size_t)(b * S_ + q0 + row) * (size_t)D_);
    char* dst = smem + OFF_Q + row * QPITCH;
    #pragma unroll
    for (int i = 0; i < 32; ++i) {
      const int c = csub + i * 32;
      v4f v = __builtin_nontemporal_load((const v4f*)(src + c));
      unsigned long long p = (unsigned long long)pk2(v.x, v.y)
                           | ((unsigned long long)pk2(v.z, v.w) << 32);
      *(unsigned long long*)(dst + c * 2) = p;
    }
  }

  // per-row online-softmax state lives in lanes 0..31 of wave 0
  float mrun = -3.0e38f, lrun = 0.f;
  bool qpad = false; int qIdx = 0;
  if (tid < 32) {
    qIdx = q0 + tid;
    qpad = (maskg[(size_t)b * S_ + qIdx] == 0);
  }

  v8f acc[16];
  const v8f vzero = {0.f,0.f,0.f,0.f,0.f,0.f,0.f,0.f};
  #pragma unroll
  for (int t = 0; t < 16; ++t) acc[t] = vzero;

  const int nkb = q0 / BN + 1;            // causal: key blocks up to q block
  for (int jb = 0; jb < nkb; ++jb) {
    const int j0 = jb * BN;

    // ---- fill K tile: 8x8 register-blocked transpose.
    //      512 tiles (4 key-chunks x 128 d-chunks); thread -> tiles tid, tid+256.
    //      Row-major copy (B of QK^T) + transposed copy (B of PV), all b128 LDS ops.
    #pragma unroll
    for (int it = 0; it < 2; ++it) {
      const int tau = tid + it * 256;
      const int kt  = tau & 3;            // 8-key chunk
      const int dt  = tau >> 2;           // 8-float d chunk
      const float* src = kg + ((size_t)(b * S_ + j0 + kt * 8) * (size_t)D_) + dt * 8;
      unsigned h[8][4];
      #pragma unroll
      for (int r = 0; r < 8; ++r) {
        v4f a  = *(const v4f*)(src + (size_t)r * D_);
        v4f c4 = *(const v4f*)(src + (size_t)r * D_ + 4);
        h[r][0] = pk2(a.x, a.y);
        h[r][1] = pk2(a.z, a.w);
        h[r][2] = pk2(c4.x, c4.y);
        h[r][3] = pk2(c4.z, c4.w);
        *(uint4*)(smem + OFF_K + (kt*8 + r) * KPITCH + dt * 16) =
            make_uint4(h[r][0], h[r][1], h[r][2], h[r][3]);
      }
      // in-register 8x8 bf16 transpose via v_perm_b32 (2 perms per 2x2 block)
      #pragma unroll
      for (int c = 0; c < 8; ++c) {
        unsigned o0, o1, o2, o3;
        const unsigned sel = (c & 1) ? 0x07060302u : 0x05040100u;
        o0 = __builtin_amdgcn_perm(h[1][c >> 1], h[0][c >> 1], sel);
        o1 = __builtin_amdgcn_perm(h[3][c >> 1], h[2][c >> 1], sel);
        o2 = __builtin_amdgcn_perm(h[5][c >> 1], h[4][c >> 1], sel);
        o3 = __builtin_amdgcn_perm(h[7][c >> 1], h[6][c >> 1], sel);
        *(uint4*)(smem + OFF_KT + (dt*8 + c) * KTPITCH + kt * 16) =
            make_uint4(o0, o1, o2, o3);
      }
      if (jb + 1 < nkb)   // global_prefetch_b8 of next K block
        __builtin_prefetch(kg + ((size_t)(b * S_ + j0 + BN + kt * 8) * (size_t)D_ + dt * 8), 0, 1);
    }
    __syncthreads();

    // ---- phase A: partial S = Q . K^T  (wave = (qb, kb, D-half)) ----
    {
      const int qb = w & 1, kb = (w >> 1) & 1, dh = w >> 2;
      const char* Abase = smem + OFF_Q + (qb*16 + lrow) * QPITCH + dh*1024 + (hih ? 16 : 0);
      const char* Bbase = smem + OFF_K + (kb*16 + lrow) * KPITCH + dh*1024 + (hih ? 16 : 0);
      v8f s = vzero;
      #pragma unroll
      for (int c = 0; c < 16; ++c) {
        v8bf alo = *(const v8bf*)(Abase + c*64);
        v8bf ahi = *(const v8bf*)(Abase + c*64 + 32);
        v8bf blo = *(const v8bf*)(Bbase + c*64);
        v8bf bhi = *(const v8bf*)(Bbase + c*64 + 32);
        s = __builtin_amdgcn_wmma_f32_16x16x32_bf16(false, cat8(alo, ahi),
                false, cat8(blo, bhi), (short)0, s, false, false);
      }
      char* Sb = smem + OFF_SP + dh * 4096;
      #pragma unroll
      for (int r = 0; r < 8; ++r) {
        const int q = qb*16 + r + (hih ? 8 : 0);
        *(float*)(Sb + (q*32 + kb*16 + lrow) * 4) = s[r];
      }
    }
    __syncthreads();

    // ---- phase B: online softmax, one row per lane of wave 0 ----
    if (tid < 32) {
      float sv[32];
      float bmax = -3.0e38f;
      const char* S0 = smem + OFF_SP + tid * 128;
      #pragma unroll
      for (int j = 0; j < 32; ++j) {
        float v = *(const float*)(S0 + j*4) + *(const float*)(S0 + 4096 + j*4);
        v *= scaleV;
        const int kIdx = j0 + j;
        const bool m = qpad || (maskg[(size_t)b * S_ + kIdx] == 0) || (kIdx > qIdx);
        if (m) v -= 1.0e9f;               // scores - NEG*mask, as in reference
        sv[j] = v;
        bmax = fmaxf(bmax, v);
      }
      const float mnew  = fmaxf(mrun, bmax);
      const float alpha = __expf(mrun - mnew);
      float lsum = 0.f;
      char* Pb = smem + OFF_P + tid * 80;
      #pragma unroll
      for (int j = 0; j < 32; ++j) {
        const float e = __expf(sv[j] - mnew);
        lsum += e;
        *(unsigned short*)(Pb + j*2) = f2bf(e);
      }
      lrun = lrun * alpha + lsum;
      mrun = mnew;
      *(float*)(smem + OFF_ALPHA + tid * 4) = alpha;
    }
    __syncthreads();

    // ---- phase C: O = alpha*O + P . K  (wave = (qb, 256-wide D slice)) ----
    {
      const int qb = w & 1, dq = w >> 1;
      const v8f af = *(const v8f*)(smem + OFF_ALPHA + (qb*16 + (hih ? 8 : 0)) * 4);
      const char* Pp = smem + OFF_P + (qb*16 + lrow) * 80 + (hih ? 16 : 0);
      const v16bf aP = cat8(*(const v8bf*)(Pp), *(const v8bf*)(Pp + 32));
      #pragma unroll
      for (int t = 0; t < 16; ++t) {
        const int d = dq*256 + t*16 + lrow;
        const char* Kp = smem + OFF_KT + d * KTPITCH + (hih ? 16 : 0);
        const v16bf bK = cat8(*(const v8bf*)(Kp), *(const v8bf*)(Kp + 32));
        acc[t] = acc[t] * af;             // flash rescale
        acc[t] = __builtin_amdgcn_wmma_f32_16x16x32_bf16(false, aP, false, bK,
                                                         (short)0, acc[t], false, false);
      }
    }
    __syncthreads();
  }

  // ---- epilogue: normalize by 1/l and store (non-temporal: never re-read) ----
  if (tid < 32) *(float*)(smem + OFF_LINV + tid * 4) = 1.0f / lrun;
  __syncthreads();
  {
    const int qb = w & 1, dq = w >> 1;
    const v8f lf = *(const v8f*)(smem + OFF_LINV + (qb*16 + (hih ? 8 : 0)) * 4);
    #pragma unroll
    for (int t = 0; t < 16; ++t) {
      #pragma unroll
      for (int r = 0; r < 8; ++r) {
        const size_t o = ((size_t)(b * S_) + q0 + qb*16 + r + (hih ? 8 : 0)) * (size_t)D_
                       + (size_t)(dq*256 + t*16 + lrow);
        __builtin_nontemporal_store(acc[t][r] * lf[r], &outg[o]);
      }
    }
  }
}

extern "C" void kernel_launch(void* const* d_in, const int* in_sizes, int n_in,
                              void* d_out, int out_size, void* d_ws, size_t ws_size,
                              hipStream_t stream) {
  (void)in_sizes; (void)n_in; (void)out_size; (void)d_ws; (void)ws_size;
  const float*          q     = (const float*)d_in[0];
  const float*          k     = (const float*)d_in[1];
  const unsigned char*  mask  = (const unsigned char*)d_in[2];  // jax bool = 1 byte
  const float*          scale = (const float*)d_in[3];
  float*                out   = (float*)d_out;
  dim3 grid(S_ / BM, B_);
  fa_fwd_kernel<<<grid, 256, 0, stream>>>(q, k, mask, scale, out);
}
